// DynamicsModel_72799695667587
// MI455X (gfx1250) — matrix operand order repaired and checked by reading, not compiled, not run on hardware
//
#include <hip/hip_runtime.h>
#include <hip/hip_bf16.h>
#include <math.h>

// ---------------- model dims ----------------
#define BB 4
#define NN 128
#define TT 64
#define DD 768
#define HEADS 8
#define DH 96            // DD / HEADS
#define D4 3072          // FFN hidden
#define L_ATTN 4
#define M_SSM 4
#define D_INNER 1536
#define D_STATE 64
#define HEADDIM 64
#define NHEADS 24
#define D_CONV 4
#define CONV_DIM 1664    // D_INNER + 2*D_STATE
#define IN_PROJ 3224     // 2*D_INNER + 2*D_STATE + NHEADS
#define NTOK 32768       // B*N*T == B*T*N rows for every GEMM

typedef __attribute__((ext_vector_type(16))) __bf16 v16bf;
typedef __attribute__((ext_vector_type(8)))  float  v8f;

__device__ __forceinline__ float siluf(float x) { return x / (1.0f + expf(-x)); }

// =====================================================================
// WMMA GEMM:  C = epilogue(A[M,K] * W[K,N] (+bias) (+residual))
// Block tile 128x128, BK=32, 8 waves in 4(m) x 2(n); each wave owns a 32x64
// sub-tile -> 8x v_wmma_f32_16x16x32_bf16 per K-step.
//
// Double-buffered LDS pipeline: global loads for tile kt+1 are issued BEFORE
// the 8 WMMAs on tile kt; cvt+ds_store for kt+1 land after the WMMAs, so the
// s_wait_loadcnt is hidden behind the matrix math.
//
// Fragment-friendly LDS layouts (every fragment = one contiguous 32B read):
//  - As[row][k]: K kept contiguous (lane half h reads k = 16h..16h+15) per 64B row.
//  - Bs[n][k]: stored transposed.
// act: 0=none 1=gelu 2=silu
// =====================================================================
#define BM 128
#define BN 128
#define BK 32

__global__ __launch_bounds__(256) void gemm_bf16_k(
    const float* __restrict__ A, const float* __restrict__ W,
    const float* __restrict__ bias, const float* __restrict__ res,
    float* __restrict__ C, int M, int K, int N, int act)
{
  __shared__ __bf16 As[2][BM][BK];   // 2 x 8 KB
  __shared__ __bf16 Bs[2][BN][BK];   // 2 x 8 KB (transposed: [n][k])
  const int tid  = threadIdx.x;
  const int wave = tid >> 5;
  const int lane = tid & 31;
  const int m0 = blockIdx.y * BM;
  const int n0 = blockIdx.x * BN;
  const int wm = (wave & 3) * 32;
  const int wn = (wave >> 2) * 64;
  const bool fullN = (n0 + BN <= N);
  const int ksteps = K / BK;

  // fixed per-thread staging coordinates: 1024 float4 per tile, 4 per thread
  int ar[4], ac[4], acp[4], br[4], bc[4];
#pragma unroll
  for (int u = 0; u < 4; ++u) {
    const int i = tid + u * 256;
    ar[u] = i >> 3;            // A: 8 float4 per row
    ac[u] = (i & 7) * 4;
    // K kept contiguous (matches Bs k order)
    acp[u] = ac[u];
    br[u] = i >> 5;            // B: 32 float4 per row
    bc[u] = (i & 31) * 4;
  }

  const v8f vzero = {0.f,0.f,0.f,0.f,0.f,0.f,0.f,0.f};
  v8f acc[2][4];
#pragma unroll
  for (int mt = 0; mt < 2; ++mt)
#pragma unroll
    for (int nt = 0; nt < 4; ++nt) acc[mt][nt] = vzero;

  float4 areg[4], breg[4];

  auto load_tile = [&](int k0) {
#pragma unroll
    for (int u = 0; u < 4; ++u)
      areg[u] = *(const float4*)&A[(size_t)(m0 + ar[u]) * K + k0 + ac[u]];
    if (fullN) {
#pragma unroll
      for (int u = 0; u < 4; ++u)
        breg[u] = *(const float4*)&W[(size_t)(k0 + br[u]) * N + n0 + bc[u]];
    } else {  // ragged last N-block (in_proj N=3224): guarded scalar loads
#pragma unroll
      for (int u = 0; u < 4; ++u) {
        const size_t base = (size_t)(k0 + br[u]) * N;
        const int n = n0 + bc[u];
        float4 f = {0.f, 0.f, 0.f, 0.f};
        if (n + 0 < N) f.x = W[base + n + 0];
        if (n + 1 < N) f.y = W[base + n + 1];
        if (n + 2 < N) f.z = W[base + n + 2];
        if (n + 3 < N) f.w = W[base + n + 3];
        breg[u] = f;
      }
    }
  };

  auto store_tile = [&](int buf) {
#pragma unroll
    for (int u = 0; u < 4; ++u) {
      union { __bf16 h[4]; uint2 uu; } pk;
      pk.h[0] = (__bf16)areg[u].x; pk.h[1] = (__bf16)areg[u].y;
      pk.h[2] = (__bf16)areg[u].z; pk.h[3] = (__bf16)areg[u].w;
      *(uint2*)&As[buf][ar[u]][acp[u]] = pk.uu;
    }
#pragma unroll
    for (int u = 0; u < 4; ++u) {
      Bs[buf][bc[u] + 0][br[u]] = (__bf16)breg[u].x;
      Bs[buf][bc[u] + 1][br[u]] = (__bf16)breg[u].y;
      Bs[buf][bc[u] + 2][br[u]] = (__bf16)breg[u].z;
      Bs[buf][bc[u] + 3][br[u]] = (__bf16)breg[u].w;
    }
  };

  // prologue: tile 0 -> buffer 0
  load_tile(0);
  store_tile(0);
  __syncthreads();

  const int half = lane >> 4;
  const int l15  = lane & 15;

  for (int kt = 0; kt < ksteps; ++kt) {
    const int cur = kt & 1;
    const bool havenext = (kt + 1 < ksteps);
    if (havenext) load_tile((kt + 1) * BK);   // issue global loads early

    v16bf af[2], bfv[4];
#pragma unroll
    for (int mt = 0; mt < 2; ++mt)
      af[mt] = *(const v16bf*)&As[cur][wm + mt * 16 + l15][half * 16];
#pragma unroll
    for (int nt = 0; nt < 4; ++nt)
      bfv[nt] = *(const v16bf*)&Bs[cur][wn + nt * 16 + l15][half * 16];
#pragma unroll
    for (int mt = 0; mt < 2; ++mt)
#pragma unroll
      for (int nt = 0; nt < 4; ++nt)
        acc[mt][nt] = __builtin_amdgcn_wmma_f32_16x16x32_bf16(
            false, af[mt], false, bfv[nt], (short)0, acc[mt][nt], false, false);

    if (havenext) store_tile(cur ^ 1);        // waits for loads after WMMAs
    __syncthreads();
  }

  // epilogue: C/D layout -> VGPR r: lanes0-15 M=r, lanes16-31 M=8+r; N=lane&15
  const int mh = (lane >> 4) * 8;
#pragma unroll
  for (int mt = 0; mt < 2; ++mt) {
#pragma unroll
    for (int nt = 0; nt < 4; ++nt) {
      const int col = n0 + wn + nt * 16 + l15;
      if (col >= N) continue;
#pragma unroll
      for (int r = 0; r < 8; ++r) {
        const int row = m0 + wm + mt * 16 + mh + r;
        float v = acc[mt][nt][r];
        if (bias) v += bias[col];
        if (act == 1) {  // tanh-approx GELU (jax.nn.gelu default)
          const float u = 0.7978845608028654f * (v + 0.044715f * v * v * v);
          v = 0.5f * v * (1.0f + tanhf(u));
        } else if (act == 2) {
          v = siluf(v);
        }
        if (res) v += res[(size_t)row * N + col];
        C[(size_t)row * N + col] = v;
      }
    }
  }
}

// =====================================================================
// RMSNorm (optionally gated by silu(gate)):  out = u * rsqrt(mean(u^2)+eps) * w
// one block per row
// =====================================================================
__global__ __launch_bounds__(256) void rmsnorm_k(
    const float* __restrict__ x, int xstride,
    const float* __restrict__ gate, int gstride,
    const float* __restrict__ w, float* __restrict__ out, int Dd)
{
  const int row = blockIdx.x;
  const float* xr = x + (size_t)row * xstride;
  const float* gr = gate ? gate + (size_t)row * gstride : nullptr;
  __shared__ float sd[256];
  const int tid = threadIdx.x;
  float s = 0.f;
  for (int c = tid; c < Dd; c += 256) {
    float v = xr[c];
    if (gr) v *= siluf(gr[c]);
    s += v * v;
  }
  sd[tid] = s; __syncthreads();
  for (int st = 128; st > 0; st >>= 1) {
    if (tid < st) sd[tid] += sd[tid + st];
    __syncthreads();
  }
  const float inv = rsqrtf(sd[0] / (float)Dd + 1e-5f);
  for (int c = tid; c < Dd; c += 256) {
    float v = xr[c];
    if (gr) v *= siluf(gr[c]);
    out[(size_t)row * Dd + c] = v * inv * w[c];
  }
}

// =====================================================================
// Attention core: one block = (sequence, head); 64 threads = 64 query rows.
// K/V staged in LDS (48 KB); q row in registers; full softmax per row.
// =====================================================================
__global__ __launch_bounds__(64) void attn_k(
    const float* __restrict__ q, const float* __restrict__ k,
    const float* __restrict__ v, float* __restrict__ o)
{
  const int seq = blockIdx.x, head = blockIdx.y;
  const size_t base = (size_t)seq * TT * DD + head * DH;
  __shared__ float ks[TT][DH], vs[TT][DH];
  const int tid = threadIdx.x;
  for (int i = tid; i < TT * DH; i += 64) {
    const int r = i / DH, c = i % DH;
    ks[r][c] = k[base + (size_t)r * DD + c];
    vs[r][c] = v[base + (size_t)r * DD + c];
  }
  __syncthreads();
  float qreg[DH];
#pragma unroll
  for (int c = 0; c < DH; ++c) qreg[c] = q[base + (size_t)tid * DD + c];

  const float scale = 0.10206207261596575f;  // 1/sqrt(96)
  float sc[TT];
  float mx = -3.4e38f;
  for (int j = 0; j < TT; ++j) {
    float s = 0.f;
    for (int c = 0; c < DH; ++c) s += qreg[c] * ks[j][c];
    s *= scale;
    sc[j] = s;
    mx = fmaxf(mx, s);
  }
  float sum = 0.f;
  for (int j = 0; j < TT; ++j) { sc[j] = expf(sc[j] - mx); sum += sc[j]; }
  const float inv = 1.f / sum;
  for (int c = 0; c < DH; ++c) {
    float a = 0.f;
    for (int j = 0; j < TT; ++j) a += sc[j] * vs[j][c];
    o[base + (size_t)tid * DD + c] = a * inv;
  }
}

// =====================================================================
// Depthwise causal conv (width 4) over xBC slice of zxbcdt, + bias + silu
// =====================================================================
__global__ __launch_bounds__(256) void conv_silu_k(
    const float* __restrict__ zxbcdt, const float* __restrict__ cw,
    const float* __restrict__ cb, float* __restrict__ xc)
{
  const long idx = (long)blockIdx.x * 256 + threadIdx.x;
  const long total = (long)NTOK * CONV_DIM;
  if (idx >= total) return;
  const int  c    = (int)(idx % CONV_DIM);
  const long rowg = idx / CONV_DIM;
  const int  l    = (int)(rowg % NN);
  const long s    = rowg / NN;
  float acc = cb[c];
#pragma unroll
  for (int kk = 0; kk < D_CONV; ++kk) {
    const int t = l + kk - (D_CONV - 1);
    if (t >= 0)
      acc += cw[kk * CONV_DIM + c] *
             zxbcdt[((size_t)(s * NN + t)) * IN_PROJ + D_INNER + c];
  }
  xc[idx] = siluf(acc);
}

// =====================================================================
// SSM selective scan: one block = (seq, head); 64 threads each own one
// headdim p and keep the 64-wide state row in registers. B/C staged in LDS.
// =====================================================================
__global__ __launch_bounds__(64) void ssm_scan_k(
    const float* __restrict__ xc, const float* __restrict__ zxbcdt,
    const float* __restrict__ dt_bias, const float* __restrict__ A_log,
    const float* __restrict__ D_skip, float* __restrict__ y,
    float* __restrict__ h_out)
{
  const int s = blockIdx.x;    // 0..255 (B*T)
  const int h = blockIdx.y;    // 0..23
  const int p = threadIdx.x;   // 0..63
  const float A   = -expf(A_log[h]);
  const float db  = dt_bias[h];
  const float dsk = D_skip[h];
  float hs[D_STATE];
#pragma unroll
  for (int i = 0; i < D_STATE; ++i) hs[i] = 0.f;
  __shared__ float Bsm[D_STATE], Csm[D_STATE];
  for (int l = 0; l < NN; ++l) {
    const size_t row = (size_t)s * NN + l;
    const float* xr = xc + row * CONV_DIM;
    Bsm[p] = xr[D_INNER + p];
    Csm[p] = xr[D_INNER + D_STATE + p];
    __syncthreads();
    const float dtr = zxbcdt[row * IN_PROJ + D_INNER + CONV_DIM + h] + db;
    const float dt  = (dtr > 20.f) ? dtr : log1pf(expf(dtr));
    const float dA  = expf(dt * A);
    const float xv  = xr[h * HEADDIM + p];
    const float dx  = dt * xv;
    float yv = 0.f;
    for (int ss2 = 0; ss2 < D_STATE; ++ss2) {
      hs[ss2] = hs[ss2] * dA + dx * Bsm[ss2];
      yv += hs[ss2] * Csm[ss2];
    }
    y[row * D_INNER + h * HEADDIM + p] = yv + dsk * xv;
    __syncthreads();
  }
  if (h_out) {
    float* hp = h_out + (((size_t)s * NHEADS + h) * HEADDIM + p) * D_STATE;
    for (int ss2 = 0; ss2 < D_STATE; ++ss2) hp[ss2] = hs[ss2];
  }
}

// =====================================================================
// Row permutes: mode 0: (B,N,T,D)->(B,T,N,D);  mode 1: (B,T,N,D)->(B,N,T,D)
// =====================================================================
__global__ __launch_bounds__(256) void permute_k(
    const float* __restrict__ src, float* __restrict__ dst, int mode)
{
  const int r = blockIdx.x;
  int drow;
  if (mode == 0) {
    const int b = r / (NN * TT); const int rem = r % (NN * TT);
    const int n = rem / TT;      const int t = rem % TT;
    drow = (b * TT + t) * NN + n;
  } else {
    const int b = r / (TT * NN); const int rem = r % (TT * NN);
    const int t = rem / NN;      const int n = rem % NN;
    drow = (b * NN + n) * TT + t;
  }
  const float* sr = src + (size_t)r * DD;
  float* dr = dst + (size_t)drow * DD;
  for (int c = threadIdx.x; c < DD; c += 256) dr[c] = sr[c];
}

// =====================================================================
extern "C" void kernel_launch(void* const* d_in, const int* in_sizes, int n_in,
                              void* d_out, int out_size, void* d_ws, size_t ws_size,
                              hipStream_t stream) {
  const float* precepts     = (const float*)d_in[0];
  const float* actions      = (const float*)d_in[1];
  const float* attn_norm1_w = (const float*)d_in[2];
  const float* Wq           = (const float*)d_in[3];
  const float* Wk           = (const float*)d_in[4];
  const float* Wv           = (const float*)d_in[5];
  const float* Wo           = (const float*)d_in[6];
  const float* attn_norm2_w = (const float*)d_in[7];
  const float* ffn_w1       = (const float*)d_in[8];
  const float* ffn_b1       = (const float*)d_in[9];
  const float* ffn_w2       = (const float*)d_in[10];
  const float* ffn_b2       = (const float*)d_in[11];
  const float* ssm_norm_w   = (const float*)d_in[12];
  const float* in_proj_w    = (const float*)d_in[13];
  const float* conv_w       = (const float*)d_in[14];
  const float* conv_b       = (const float*)d_in[15];
  const float* dt_bias      = (const float*)d_in[16];
  const float* A_log        = (const float*)d_in[17];
  const float* D_skip       = (const float*)d_in[18];
  const float* mamba_norm_w = (const float*)d_in[19];
  const float* out_proj_w   = (const float*)d_in[20];
  const float* proj_w       = (const float*)d_in[21];
  const float* proj_b       = (const float*)d_in[22];

  const size_t TD = (size_t)NTOK * DD;
  float* ws  = (float*)d_ws;
  float* p   = ws; ws += TD;
  float* tn  = ws; ws += TD;
  float* qb  = ws; ws += TD;
  float* kb  = ws; ws += TD;
  float* vb  = ws; ws += TD;
  float* ob  = ws; ws += TD;
  float* z   = ws; ws += TD;
  float* big = ws; ws += (size_t)NTOK * IN_PROJ;   // zxbcdt / ffn hidden
  float* xc  = ws; ws += (size_t)NTOK * CONV_DIM;
  float* yv  = ws; ws += (size_t)NTOK * D_INNER;
  float* yn  = ws; ws += (size_t)NTOK * D_INNER;

  hipMemcpyAsync(p, precepts, TD * sizeof(float), hipMemcpyDeviceToDevice, stream);

  const dim3 blk(256);
  const dim3 gN(DD / BN, NTOK / BM);                 // N = 768  -> 6 x-blocks
  const dim3 gH(D4 / BN, NTOK / BM);                 // N = 3072 -> 24
  const dim3 gIP((IN_PROJ + BN - 1) / BN, NTOK / BM);// N = 3224 -> 26 (ragged)

  // ---------------- attention / FFN stack ----------------
  for (int i = 0; i < L_ATTN; ++i) {
    const bool cross = (((i + 1) & 1) == 0);
    rmsnorm_k<<<NTOK, 256, 0, stream>>>(p, DD, nullptr, 0, attn_norm1_w + i * DD, tn, DD);
    const float* kvsrc = cross ? actions : tn;
    gemm_bf16_k<<<gN, blk, 0, stream>>>(tn,    Wq + (size_t)i * DD * DD, nullptr, nullptr, qb, NTOK, DD, DD, 0);
    gemm_bf16_k<<<gN, blk, 0, stream>>>(kvsrc, Wk + (size_t)i * DD * DD, nullptr, nullptr, kb, NTOK, DD, DD, 0);
    gemm_bf16_k<<<gN, blk, 0, stream>>>(kvsrc, Wv + (size_t)i * DD * DD, nullptr, nullptr, vb, NTOK, DD, DD, 0);
    attn_k<<<dim3(BB * NN, HEADS), dim3(64), 0, stream>>>(qb, kb, vb, ob);
    gemm_bf16_k<<<gN, blk, 0, stream>>>(ob, Wo + (size_t)i * DD * DD, nullptr, p, p, NTOK, DD, DD, 0);
    rmsnorm_k<<<NTOK, 256, 0, stream>>>(p, DD, nullptr, 0, attn_norm2_w + i * DD, tn, DD);
    gemm_bf16_k<<<gH, blk, 0, stream>>>(tn,  ffn_w1 + (size_t)i * DD * D4, ffn_b1 + i * D4, nullptr, big, NTOK, DD, D4, 1);
    gemm_bf16_k<<<gN, blk, 0, stream>>>(big, ffn_w2 + (size_t)i * D4 * DD, ffn_b2 + i * DD, p,       p,   NTOK, D4, DD, 0);
  }

  // (B,N,T,D) -> (B,T,N,D)
  permute_k<<<NTOK, 256, 0, stream>>>(p, z, 0);

  // ---------------- Mamba2 blocks ----------------
  float* hlast_out = (float*)d_out + TD;
  for (int m = 0; m < M_SSM; ++m) {
    rmsnorm_k<<<NTOK, 256, 0, stream>>>(z, DD, nullptr, 0, ssm_norm_w + m * DD, tn, DD);
    gemm_bf16_k<<<gIP, blk, 0, stream>>>(tn, in_proj_w + (size_t)m * DD * IN_PROJ,
                                         nullptr, nullptr, big, NTOK, DD, IN_PROJ, 0);
    const long convTotal = (long)NTOK * CONV_DIM;
    conv_silu_k<<<(convTotal + 255) / 256, 256, 0, stream>>>(
        big, conv_w + (size_t)m * D_CONV * CONV_DIM, conv_b + m * CONV_DIM, xc);
    ssm_scan_k<<<dim3(BB * TT, NHEADS), dim3(64), 0, stream>>>(
        xc, big, dt_bias + m * NHEADS, A_log + m * NHEADS, D_skip + m * NHEADS,
        yv, (m == M_SSM - 1) ? hlast_out : nullptr);
    // gated rmsnorm: u = y * silu(z-gate cols of zxbcdt), normalized over D_INNER
    rmsnorm_k<<<NTOK, 256, 0, stream>>>(yv, D_INNER, big, IN_PROJ,
                                        mamba_norm_w + m * D_INNER, yn, D_INNER);
    gemm_bf16_k<<<gN, blk, 0, stream>>>(yn, out_proj_w + (size_t)m * D_INNER * DD,
                                        nullptr, z, z, NTOK, D_INNER, DD, 0);
  }

  // ---------------- final projection + output layout ----------------
  gemm_bf16_k<<<gN, blk, 0, stream>>>(z, proj_w, proj_b, nullptr, tn, NTOK, DD, DD, 0);
  permute_k<<<NTOK, 256, 0, stream>>>(tn, (float*)d_out, 1);
}
